// GAT_78400333021695
// MI455X (gfx1250) — matrix operand order repaired and checked by reading, not compiled, not run on hardware
//
#include <hip/hip_runtime.h>

#define NN     50000
#define EE     800000
#define ETOT   (EE + NN)
#define HIDD   128
#define NHEAD  4
#define CH     32
#define GG     512
#define LAT    64
#define NEG    0.2f
#define BNEPS  1e-5f

#define LDS_STRIDE 136   // 128 + 8 halves pad: conflict-free rows, 16B-aligned chunks

typedef __attribute__((ext_vector_type(16))) _Float16 v16h_t;
typedef __attribute__((ext_vector_type(8)))  _Float16 v8h_t;
typedef __attribute__((ext_vector_type(4)))  _Float16 v4h_t;
typedef __attribute__((ext_vector_type(8)))  float    v8f_t;

// ---------------------------------------------------------------------------
// WMMA GEMM: hW[N,128] = act(X[N,128]) @ W[128,128]
// act = leaky_relu(0.2) when apply_leaky!=0 (fused activation of prev layer).
// W staged TRANSPOSED in LDS (Wt[col][k]) so B-fragments are contiguous
// ds_load_b128 pairs. Rows padded to 272B so A/B fragment loads are
// bank-conflict free. A tile staged as float4 -> packed-half b64 stores.
// Persistent blocks; next tile prefetched.
// ---------------------------------------------------------------------------
__global__ void k_gat_gemm(const float* __restrict__ X, const float* __restrict__ W,
                           float* __restrict__ hW, int apply_leaky) {
  __shared__ _Float16 Wt[128][LDS_STRIDE];  // ~34 KB, Wt[col][k] = W[k][col]
  __shared__ _Float16 As[16][LDS_STRIDE];   // ~4.3 KB
  const int tid = threadIdx.x;              // 0..255

  // one-time W staging: coalesced float4 loads, transposed b16 scatter stores
  for (int i = tid * 4; i < 128 * 128; i += 1024) {
    const float4 w4 = *(const float4*)(W + i);
    const int k = i >> 7, c = i & 127;
    Wt[c + 0][k] = (_Float16)w4.x;
    Wt[c + 1][k] = (_Float16)w4.y;
    Wt[c + 2][k] = (_Float16)w4.z;
    Wt[c + 3][k] = (_Float16)w4.w;
  }

  const int wave  = tid >> 5;
  const int lane  = tid & 31;
  const int col0  = wave * 16;
  const int r     = lane & 15;
  const int khalf = lane >> 4;              // 0 or 1
  const int col   = col0 + r;
  const int ntiles = NN / 16;               // 3125, exact

  for (int tile = blockIdx.x; tile < ntiles; tile += gridDim.x) {
    const int row0 = tile * 16;
    __syncthreads();  // Wt visible (1st iter) / previous As readers done
    // A tile is contiguous in X: vectorized 2-iteration copy (b128 -> b64)
    const float* __restrict__ Xt = X + (size_t)row0 * HIDD;
    for (int i = tid * 4; i < 16 * 128; i += 1024) {
      float4 v4 = *(const float4*)(Xt + i);
      if (apply_leaky) {
        v4.x = v4.x > 0.f ? v4.x : NEG * v4.x;
        v4.y = v4.y > 0.f ? v4.y : NEG * v4.y;
        v4.z = v4.z > 0.f ? v4.z : NEG * v4.z;
        v4.w = v4.w > 0.f ? v4.w : NEG * v4.w;
      }
      v4h_t hv = { (_Float16)v4.x, (_Float16)v4.y, (_Float16)v4.z, (_Float16)v4.w };
      *(v4h_t*)&As[i >> 7][i & 127] = hv;
    }
    // prefetch next tile's A rows (16 rows * 512B = 64 cachelines)
    const int ntile = tile + gridDim.x;
    if (ntile < ntiles && tid < 64)
      __builtin_prefetch(X + (size_t)(ntile * 16 + (tid >> 2)) * HIDD + (tid & 3) * 32, 0, 1);
    __syncthreads();

    v8f_t acc = {};
    #pragma unroll
    for (int kk = 0; kk < 4; ++kk) {
      const int kb = kk * 32;
      // A: lane holds row r; elems 0..15 -> K = kb + khalf*16 + i (contiguous)
      v8h_t alo = *(const v8h_t*)&As[r][kb + khalf * 16];
      v8h_t ahi = *(const v8h_t*)&As[r][kb + khalf * 16 + 8];
      v16h_t a = __builtin_shufflevector(alo, ahi, 0, 1, 2, 3, 4, 5, 6, 7,
                                         8, 9, 10, 11, 12, 13, 14, 15);
      // B: lane holds col; elems 0..15 -> K = kb + khalf*16 + i (contiguous in Wt)
      v8h_t blo = *(const v8h_t*)&Wt[col][kb + khalf * 16];
      v8h_t bhi = *(const v8h_t*)&Wt[col][kb + khalf * 16 + 8];
      v16h_t b = __builtin_shufflevector(blo, bhi, 0, 1, 2, 3, 4, 5, 6, 7,
                                         8, 9, 10, 11, 12, 13, 14, 15);
      acc = __builtin_amdgcn_wmma_f32_16x16x32_f16(
          false, a, false, b, (short)0, acc, false, false);
    }
    // C/D layout: VGPR v -> M = v + (lane>=16 ? 8:0), N = lane&15.
    // Single base pointer; per-row offsets (v*512B) fold into store immediates.
    float* __restrict__ pout = hW + (size_t)(row0 + khalf * 8) * HIDD + col;
    #pragma unroll
    for (int v = 0; v < 8; ++v)
      pout[v * HIDD] = acc[v];
  }
}

// ---------------------------------------------------------------------------
// Per-node attention logits + init of segment-max (ordered-int -inf) & denom
// ---------------------------------------------------------------------------
__device__ __forceinline__ int f2ord(float f) {
  int i = __float_as_int(f);
  return i >= 0 ? i : (i ^ 0x7fffffff);
}
__device__ __forceinline__ float ord2f(int i) {
  return __int_as_float(i >= 0 ? i : (i ^ 0x7fffffff));
}

__global__ void k_alpha(const float* __restrict__ hW, const float* __restrict__ a_s,
                        const float* __restrict__ a_d, float* __restrict__ asrc,
                        float* __restrict__ adst, int* __restrict__ m_i,
                        float* __restrict__ denom) {
  int n = blockIdx.x * blockDim.x + threadIdx.x;
  if (n >= NN) return;
  const float* hr = hW + (size_t)n * HIDD;
  #pragma unroll
  for (int h = 0; h < NHEAD; ++h) {
    float ss = 0.f, sd = 0.f;
    #pragma unroll 8
    for (int c = 0; c < CH; ++c) {
      float v = hr[h * CH + c];
      ss += v * a_s[h * CH + c];
      sd += v * a_d[h * CH + c];
    }
    asrc[n * NHEAD + h]  = ss;
    adst[n * NHEAD + h]  = sd;
    m_i[n * NHEAD + h]   = (int)0x80000000;  // below every ordered score
    denom[n * NHEAD + h] = 0.f;
  }
}

__global__ void k_init_out(float* __restrict__ out, const float* __restrict__ bias) {
  int i = blockIdx.x * blockDim.x + threadIdx.x;
  if (i < NN * HIDD) out[i] = bias[i & 127];
}

// edge helpers: e < EE -> real edge, else self loop (src=dst=e-EE)
__device__ __forceinline__ void edge_sd(const int* __restrict__ ei, int e, int& s, int& d) {
  if (e < EE) { s = ei[e]; d = ei[EE + e]; }
  else        { s = e - EE; d = s; }
}

__global__ void k_edge_max(const int* __restrict__ ei, const float* __restrict__ asrc,
                           const float* __restrict__ adst, int* __restrict__ m_i) {
  int e = blockIdx.x * blockDim.x + threadIdx.x;
  if (e >= ETOT) return;
  int s, d; edge_sd(ei, e, s, d);
  #pragma unroll
  for (int h = 0; h < NHEAD; ++h) {
    float sc = asrc[s * NHEAD + h] + adst[d * NHEAD + h];
    sc = sc > 0.f ? sc : NEG * sc;
    atomicMax(&m_i[d * NHEAD + h], f2ord(sc));
  }
}

__global__ void k_fix(int* __restrict__ m_i) {   // ordered-int -> float, nonfinite -> 0
  int i = blockIdx.x * blockDim.x + threadIdx.x;
  if (i >= NN * NHEAD) return;
  float v = ord2f(m_i[i]);
  ((float*)m_i)[i] = __builtin_isfinite(v) ? v : 0.f;
}

__global__ void k_edge_sum(const int* __restrict__ ei, const float* __restrict__ asrc,
                           const float* __restrict__ adst, const float* __restrict__ mf,
                           float* __restrict__ denom, float* __restrict__ alphaE) {
  int e = blockIdx.x * blockDim.x + threadIdx.x;
  if (e >= ETOT) return;
  int s, d; edge_sd(ei, e, s, d);
  #pragma unroll
  for (int h = 0; h < NHEAD; ++h) {
    float sc = asrc[s * NHEAD + h] + adst[d * NHEAD + h];
    sc = sc > 0.f ? sc : NEG * sc;
    float ex = __expf(sc - mf[d * NHEAD + h]);
    alphaE[(size_t)e * NHEAD + h] = ex;
    atomicAdd(&denom[d * NHEAD + h], ex);
  }
}

// 128 threads per edge: coalesced row gather of hW[src], coalesced atomics to out[dst]
__global__ void k_edge_aggr(const int* __restrict__ ei, const float* __restrict__ alphaE,
                            const float* __restrict__ denom, const float* __restrict__ hW,
                            float* __restrict__ out) {
  int gid = blockIdx.x * blockDim.x + threadIdx.x;
  if (gid >= ETOT * HIDD) return;
  int e = gid >> 7, k = gid & 127, h = k >> 5;
  int s, d; edge_sd(ei, e, s, d);
  const float* __restrict__ hs = hW + (size_t)s * HIDD;
  float* __restrict__ od = out + (size_t)d * HIDD;
  float a = alphaE[(size_t)e * NHEAD + h] / (denom[d * NHEAD + h] + 1e-16f);
  atomicAdd(&od[k], a * hs[k]);
}

// ---------------------------------------------------------------------------
// Pool + BN(eval) + FC
// ---------------------------------------------------------------------------
__global__ void k_zero(float* __restrict__ p, int n) {
  int i = blockIdx.x * blockDim.x + threadIdx.x;
  if (i < n) p[i] = 0.f;
}

__global__ void k_pool(const float* __restrict__ hfin, const int* __restrict__ batch,
                       float* __restrict__ g) {
  int i = blockIdx.x * blockDim.x + threadIdx.x;
  if (i >= NN * HIDD) return;
  int n = i >> 7, k = i & 127;
  float v = hfin[i];
  v = v > 0.f ? v : NEG * v;   // final layer activation fused here
  atomicAdd(&g[(size_t)batch[n] * HIDD + k], v);
}

__global__ void k_bn_fc(const float* __restrict__ g, const float* __restrict__ gamma,
                        const float* __restrict__ beta, const float* __restrict__ mean,
                        const float* __restrict__ var, const float* __restrict__ Wfc,
                        const float* __restrict__ bfc, float* __restrict__ out) {
  int t = blockIdx.x * blockDim.x + threadIdx.x;
  if (t >= GG * LAT) return;
  int gi = t >> 6, j = t & 63;
  const float* gr = g + (size_t)gi * HIDD;
  float acc = 0.f;
  for (int k = 0; k < HIDD; ++k) {
    float v = (gr[k] - mean[k]) * rsqrtf(var[k] + BNEPS) * gamma[k] + beta[k];
    acc += v * Wfc[k * LAT + j];
  }
  out[t] = acc + bfc[j];
}

// ---------------------------------------------------------------------------
extern "C" void kernel_launch(void* const* d_in, const int* in_sizes, int n_in,
                              void* d_out, int out_size, void* d_ws, size_t ws_size,
                              hipStream_t stream) {
  (void)in_sizes; (void)n_in; (void)out_size; (void)ws_size;
  const float* x     = (const float*)d_in[0];
  const int*   ei    = (const int*)  d_in[1];
  const int*   batch = (const int*)  d_in[2];
  const float* Wl[3]  = { (const float*)d_in[3],  (const float*)d_in[7],  (const float*)d_in[11] };
  const float* asl[3] = { (const float*)d_in[4],  (const float*)d_in[8],  (const float*)d_in[12] };
  const float* adl[3] = { (const float*)d_in[5],  (const float*)d_in[9],  (const float*)d_in[13] };
  const float* bl[3]  = { (const float*)d_in[6],  (const float*)d_in[10], (const float*)d_in[14] };
  const float* bn_g = (const float*)d_in[15];
  const float* bn_b = (const float*)d_in[16];
  const float* bn_m = (const float*)d_in[17];
  const float* bn_v = (const float*)d_in[18];
  const float* Wfc  = (const float*)d_in[19];
  const float* bfc  = (const float*)d_in[20];
  float* out = (float*)d_out;

  // workspace carve-out (~94 MB)
  char* ws = (char*)d_ws; size_t off = 0;
  auto carve = [&](size_t bytes) -> void* {
    void* p = ws + off; off += (bytes + 255) & ~(size_t)255; return p;
  };
  float* hW     = (float*)carve((size_t)NN * HIDD * 4);
  float* bufA   = (float*)carve((size_t)NN * HIDD * 4);
  float* bufB   = (float*)carve((size_t)NN * HIDD * 4);
  float* asrc   = (float*)carve((size_t)NN * NHEAD * 4);
  float* adst   = (float*)carve((size_t)NN * NHEAD * 4);
  int*   m_i    = (int*)  carve((size_t)NN * NHEAD * 4);
  float* denom  = (float*)carve((size_t)NN * NHEAD * 4);
  float* alphaE = (float*)carve((size_t)ETOT * NHEAD * 4);
  float* gpool  = (float*)carve((size_t)GG * HIDD * 4);

  const int TB = 256;
  const int gemm_blocks = 1024;                       // persistent, 3125 tiles total
  const int nb_node  = (NN + TB - 1) / TB;
  const int nb_nh    = (NN * HIDD + TB - 1) / TB;
  const int nb_nhd4  = (NN * NHEAD + TB - 1) / TB;
  const int nb_edge  = (ETOT + TB - 1) / TB;
  const int nb_aggr  = (ETOT * HIDD + TB - 1) / TB;

  const float* layer_in[3] = { x, bufA, bufB };
  float*       layer_out[3] = { bufA, bufB, bufA };

  for (int L = 0; L < 3; ++L) {
    k_gat_gemm<<<gemm_blocks, TB, 0, stream>>>(layer_in[L], Wl[L], hW, L > 0);
    k_alpha   <<<nb_node, TB, 0, stream>>>(hW, asl[L], adl[L], asrc, adst, m_i, denom);
    k_init_out<<<nb_nh,   TB, 0, stream>>>(layer_out[L], bl[L]);
    k_edge_max<<<nb_edge, TB, 0, stream>>>(ei, asrc, adst, m_i);
    k_fix     <<<nb_nhd4, TB, 0, stream>>>(m_i);
    k_edge_sum<<<nb_edge, TB, 0, stream>>>(ei, asrc, adst, (const float*)m_i, denom, alphaE);
    k_edge_aggr<<<nb_aggr, TB, 0, stream>>>(ei, alphaE, denom, hW, layer_out[L]);
  }

  k_zero <<<(GG * HIDD + TB - 1) / TB, TB, 0, stream>>>(gpool, GG * HIDD);
  k_pool <<<nb_nh, TB, 0, stream>>>(bufA, batch, gpool);
  k_bn_fc<<<(GG * LAT + TB - 1) / TB, TB, 0, stream>>>(gpool, bn_g, bn_b, bn_m, bn_v,
                                                       Wfc, bfc, out);
}
